// Decoder_23424751633096
// MI455X (gfx1250) — compile-verified, matrix operand order checked by reading, not run on hardware
//
#include <hip/hip_runtime.h>
#include <hip/hip_bf16.h>
#include <math.h>

// ---------------------------------------------------------------------------
// Tacotron2-style decoder scan for MI455X (gfx1250).
//  - bf16 WMMA (v_wmma_f32_16x16x32_bf16) for all recurrent GEMMs
//  - weights packed once to bf16 fused [wih|whh] layouts (fit in 192MB L2)
//  - one persistent kernel for the 500-step scan, 2 grid barriers per step
// ---------------------------------------------------------------------------

typedef __attribute__((ext_vector_type(8)))  __bf16 v8bf;
typedef __attribute__((ext_vector_type(16))) __bf16 v16bf;
typedef __attribute__((ext_vector_type(8)))  float  v8f;

#define T_DEC 500
#define BATCH 32
#define ATTD  1024
#define ED    768
#define PRED  256
#define KA    2048      // att fused K  = (PRE+E) + ATT
#define KD    2816      // dec fused K  = (DEC+E) + DEC
#define KP    1792      // proj K       = DEC+E
#define ZO    0.1f
#define NWG   32
#define TPB   256

__device__ __forceinline__ float sigf(float x) { return 1.f / (1.f + __expf(-x)); }

__device__ __forceinline__ unsigned hashu(unsigned x) {
  x ^= x >> 16; x *= 0x7feb352dU; x ^= x >> 15; x *= 0x846ca68bU; x ^= x >> 16;
  return x;
}

// ---- WMMA fragment helpers (ISA 7.12.2, 16-bit 16x32 A / 32x16 B layout) ----
// lane<16 : row = lane,    K = k0+[0..7] and k0+[16..23]
// lane>=16: row = lane-16, K = k0+[8..15] and k0+[24..31]
__device__ __forceinline__ v16bf load_frag(const __bf16* base, int ld,
                                           int row0, int col0, int lane) {
  int r = row0 + (lane & 15);
  int c = col0 + ((lane >> 4) << 3);
  const __bf16* p = base + (long)r * ld + c;
  v8bf lo = *(const v8bf*)(p);
  v8bf hi = *(const v8bf*)(p + 16);
  return __builtin_shufflevector(lo, hi, 0,1,2,3,4,5,6,7,8,9,10,11,12,13,14,15);
}

__device__ __forceinline__ v8f wmma_bf16(v16bf a, v16bf b, v8f c) {
  return __builtin_amdgcn_wmma_f32_16x16x32_bf16(false, a, false, b,
                                                 (short)0, c, false, false);
}

// Accumulate NG gate tiles (columns g*gstride + n0) over one K segment.
template <int NG>
__device__ __forceinline__ void gemm_seg(v8f* acc, int gstride,
                                         const __bf16* A, int lda, int m0,
                                         const __bf16* W, int ldw, int n0,
                                         int wcol, int klen, int lane) {
  for (int kc = 0; kc < klen; kc += 32) {
    v16bf a = load_frag(A, lda, m0, kc, lane);
#pragma unroll
    for (int g = 0; g < NG; ++g) {
      v16bf b = load_frag(W, ldw, g * gstride + n0, wcol + kc, lane);
      acc[g] = wmma_bf16(a, b, acc[g]);
    }
  }
}

// ---- grid-wide barrier (persistent kernel) ----
__device__ __forceinline__ void grid_sync(unsigned* cnt, unsigned* gen) {
  __syncthreads();
  if (threadIdx.x == 0) {
    __threadfence();
    unsigned g = *(volatile unsigned*)gen;
    unsigned a = atomicAdd(cnt, 1u);
    if (a == gridDim.x - 1) {
      *(volatile unsigned*)cnt = 0u;
      __threadfence();
      atomicAdd(gen, 1u);
    } else {
      while (*(volatile unsigned*)gen == g) { __builtin_amdgcn_s_sleep(2); }
    }
    __threadfence();
  }
  __syncthreads();
}

// ---- per-wave task bodies ----
__device__ __forceinline__ void do_att(int task, int t, int rd, int wr, int lane,
                                       const __bf16* Wa, const __bf16* pre,
                                       const __bf16* ctx, __bf16* ahb,
                                       float* ah, float* ac,
                                       const float* abih, const float* abhh) {
  int m0 = (task & 1) * 16;
  int h0 = (task >> 1) * 16;
  v8f zero = {0.f,0.f,0.f,0.f,0.f,0.f,0.f,0.f};
  v8f acc[4] = {zero, zero, zero, zero};
  const __bf16* preA = pre + (long)t * BATCH * PRED;
  const __bf16* ctxA = ctx + (long)t * BATCH * ED;
  const __bf16* ahA  = ahb + (long)rd * BATCH * ATTD;
  gemm_seg<4>(acc, ATTD, preA, PRED, m0, Wa, KA, h0, 0,         PRED, lane);
  gemm_seg<4>(acc, ATTD, ctxA, ED,   m0, Wa, KA, h0, PRED,      ED,   lane);
  gemm_seg<4>(acc, ATTD, ahA,  ATTD, m0, Wa, KA, h0, PRED + ED, ATTD, lane);
  int h = h0 + (lane & 15);
  int roff = (lane >= 16) ? 8 : 0;
#pragma unroll
  for (int r = 0; r < 8; ++r) {
    int bb = m0 + r + roff;
    long idx = (long)bb * ATTD + h;
    float gi = acc[0][r] + abih[h]          + abhh[h];
    float gf = acc[1][r] + abih[1024 + h]   + abhh[1024 + h];
    float gg = acc[2][r] + abih[2048 + h]   + abhh[2048 + h];
    float go = acc[3][r] + abih[3072 + h]   + abhh[3072 + h];
    float c0 = ac[idx], hh0 = ah[idx];
    float c2 = sigf(gf) * c0 + sigf(gi) * tanhf(gg);
    float h2 = sigf(go) * tanhf(c2);
    float cn = ZO * c0  + (1.f - ZO) * c2;   // deterministic zoneout blend
    float hn = ZO * hh0 + (1.f - ZO) * h2;
    ac[idx] = cn; ah[idx] = hn;
    ahb[(long)wr * BATCH * ATTD + idx] = (__bf16)hn;
  }
}

__device__ __forceinline__ void do_dec(int task, int t, int rd, int wr, int lane,
                                       const __bf16* Wd, const __bf16* ctx,
                                       const __bf16* ahb, __bf16* dhb,
                                       float* dh, float* dc,
                                       const float* dbih, const float* dbhh) {
  int m0 = (task & 1) * 16;
  int h0 = (task >> 1) * 16;
  v8f zero = {0.f,0.f,0.f,0.f,0.f,0.f,0.f,0.f};
  v8f acc[4] = {zero, zero, zero, zero};
  const __bf16* ahA  = ahb + (long)wr * BATCH * ATTD;   // ah_t (written this step)
  const __bf16* ctxA = ctx + (long)t * BATCH * ED;
  const __bf16* dhA  = dhb + (long)rd * BATCH * ATTD;   // dh_{t-1}
  gemm_seg<4>(acc, ATTD, ahA,  ATTD, m0, Wd, KD, h0, 0,    ATTD, lane);
  gemm_seg<4>(acc, ATTD, ctxA, ED,   m0, Wd, KD, h0, 1024, ED,   lane);
  gemm_seg<4>(acc, ATTD, dhA,  ATTD, m0, Wd, KD, h0, 1792, ATTD, lane);
  int h = h0 + (lane & 15);
  int roff = (lane >= 16) ? 8 : 0;
#pragma unroll
  for (int r = 0; r < 8; ++r) {
    int bb = m0 + r + roff;
    long idx = (long)bb * ATTD + h;
    float gi = acc[0][r] + dbih[h]          + dbhh[h];
    float gf = acc[1][r] + dbih[1024 + h]   + dbhh[1024 + h];
    float gg = acc[2][r] + dbih[2048 + h]   + dbhh[2048 + h];
    float go = acc[3][r] + dbih[3072 + h]   + dbhh[3072 + h];
    float c0 = dc[idx], hh0 = dh[idx];
    float c2 = sigf(gf) * c0 + sigf(gi) * tanhf(gg);
    float h2 = sigf(go) * tanhf(c2);
    float cn = ZO * c0  + (1.f - ZO) * c2;
    float hn = ZO * hh0 + (1.f - ZO) * h2;
    dc[idx] = cn; dh[idx] = hn;
    dhb[(long)wr * BATCH * ATTD + idx] = (__bf16)hn;
  }
}

__device__ __forceinline__ void do_proj(int task, int tp, int rdbuf, int lane,
                                        const __bf16* Wp, const __bf16* ctx,
                                        const __bf16* dhb, const float* pb,
                                        float* mel) {
  int m0 = (task & 1) * 16;
  int n0 = (task >> 1) * 16;           // n0 in [0,160)
  v8f zero = {0.f,0.f,0.f,0.f,0.f,0.f,0.f,0.f};
  v8f acc[1] = {zero};
  const __bf16* dhA  = dhb + (long)rdbuf * BATCH * ATTD;  // dh_tp
  const __bf16* ctxA = ctx + (long)tp * BATCH * ED;
  gemm_seg<1>(acc, 0, dhA,  ATTD, m0, Wp, KP, n0, 0,    ATTD, lane);
  gemm_seg<1>(acc, 0, ctxA, ED,   m0, Wp, KP, n0, ATTD, ED,   lane);
  int n = n0 + (lane & 15);
  int roff = (lane >= 16) ? 8 : 0;
#pragma unroll
  for (int r = 0; r < 8; ++r) {
    int bb = m0 + r + roff;
    float v = acc[0][r] + pb[n];
    // outs[tp][b][n] -> mel[b][n%80][2*tp + n/80]
    mel[(long)bb * 80000 + (n % 80) * 1000 + 2 * tp + n / 80] = v;
  }
}

// ---- persistent scan kernel ----
__global__ void __launch_bounds__(TPB)
scan_kernel(const __bf16* Wa, const __bf16* Wd, const __bf16* Wp,
            const __bf16* pre, const __bf16* ctx,
            const float* abih, const float* abhh,
            const float* dbih, const float* dbhh, const float* pb,
            float* ah, float* ac, float* dh, float* dc,
            __bf16* ahb, __bf16* dhb, float* mel, unsigned* sync) {
  int lane   = threadIdx.x & 31;
  int gwave  = blockIdx.x * (TPB / 32) + (threadIdx.x >> 5);
  int nwaves = gridDim.x * (TPB / 32);
  for (int t = 0; t <= T_DEC; ++t) {
    int rd = t & 1, wr = rd ^ 1;
    int natt  = (t < T_DEC) ? 128 : 0;
    int nproj = (t > 0) ? 20 : 0;
    // phase 1: attention gates + fused LSTM update, plus projection of step t-1
    for (int task = gwave; task < natt + nproj; task += nwaves) {
      if (task < natt) do_att(task, t, rd, wr, lane, Wa, pre, ctx, ahb, ah, ac, abih, abhh);
      else             do_proj(task - natt, t - 1, rd, lane, Wp, ctx, dhb, pb, mel);
    }
    grid_sync(sync, sync + 1);
    // phase 2: decoder gates + fused LSTM update
    if (t < T_DEC) {
      for (int task = gwave; task < 128; task += nwaves)
        do_dec(task, t, rd, wr, lane, Wd, ctx, ahb, dhb, dh, dc, dbih, dbhh);
    }
    grid_sync(sync, sync + 1);
  }
}

// ---- preprocessing kernels ----
__global__ void init_kernel(unsigned* sync, float* st, __bf16* ahb, __bf16* dhb) {
  int i = blockIdx.x * blockDim.x + threadIdx.x;
  int stride = gridDim.x * blockDim.x;
  if (i < 8) sync[i] = 0u;
  for (int j = i; j < 4 * BATCH * ATTD; j += stride) st[j] = 0.f;
  for (int j = i; j < 2 * BATCH * ATTD; j += stride) {
    ahb[j] = (__bf16)0.f; dhb[j] = (__bf16)0.f;
  }
}

__global__ void pack2_bf16(const float* s1, int k1, const float* s2, int k2,
                           __bf16* dst, int N) {
  int K = k1 + k2;
  long total = (long)N * K;
  long stride = (long)gridDim.x * blockDim.x;
  for (long i = blockIdx.x * (long)blockDim.x + threadIdx.x; i < total; i += stride) {
    long n = i / K; int k = (int)(i % K);
    float v = (k < k1) ? s1[n * k1 + k] : s2[n * k2 + (k - k1)];
    dst[i] = (__bf16)v;
  }
}

__global__ void ctx_kernel(const float* mem, __bf16* ctx) {
  long total = (long)T_DEC * BATCH * ED;
  long stride = (long)gridDim.x * blockDim.x;
  for (long i = blockIdx.x * (long)blockDim.x + threadIdx.x; i < total; i += stride) {
    int e = (int)(i % ED);
    long r = i / ED;
    int b = (int)(r % BATCH);
    int t = (int)(r / BATCH);
    const float* p = mem + ((long)b * 1000 + 2 * t) * ED + e;
    ctx[i] = (__bf16)(0.5f * (p[0] + p[ED]));
  }
}

__global__ void __launch_bounds__(256)
prenet_kernel(const float* dec_in, const float* w1, const float* w2, __bf16* pre) {
  __shared__ float xr[160];
  __shared__ float h1[256];
  int blk = blockIdx.x;
  int t = blk >> 5, b = blk & 31;    // t in [0,500)
  int tid = threadIdx.x;
  if (tid < 160) {
    float v = 0.f;
    if (t > 0) {
      int m = tid % 80, tt = 2 * (t - 1) + tid / 80;
      v = dec_in[((long)b * 80 + m) * 1000 + tt];
    }
    xr[tid] = v;
  }
  __syncthreads();
  {
    float s = 0.f;
    const float* w = w1 + tid * 160;
    for (int k = 0; k < 160; ++k) s = fmaf(w[k], xr[k], s);
    s = fmaxf(s, 0.f);
    unsigned hs = hashu((unsigned)(t * 32 + b) * 2654435761u + tid * 40503u + 0x9e3779b9u);
    h1[tid] = ((hs & 0xFFFFFFu) < (1u << 23)) ? s * 2.f : 0.f;  // p_drop=0.5
  }
  __syncthreads();
  {
    float s = 0.f;
    const float* w = w2 + tid * 256;
    for (int k = 0; k < 256; ++k) s = fmaf(w[k], h1[k], s);
    s = fmaxf(s, 0.f);
    unsigned hs = hashu((unsigned)(t * 32 + b) * 2246822519u + tid * 3266489u + 0x85ebca6bu);
    s = ((hs & 0xFFFFFFu) < (1u << 23)) ? s * 2.f : 0.f;
    pre[((long)(t * 32 + b)) * PRED + tid] = (__bf16)s;
  }
}

// ---------------------------------------------------------------------------
extern "C" void kernel_launch(void* const* d_in, const int* in_sizes, int n_in,
                              void* d_out, int out_size, void* d_ws, size_t ws_size,
                              hipStream_t stream) {
  (void)in_sizes; (void)n_in; (void)out_size; (void)ws_size;
  const float* memory  = (const float*)d_in[0];
  const float* dec_in  = (const float*)d_in[1];
  // d_in[2] memory_lengths: unused by the reference computation
  const float* w1      = (const float*)d_in[3];
  const float* w2      = (const float*)d_in[4];
  const float* att_wih = (const float*)d_in[5];
  const float* att_whh = (const float*)d_in[6];
  const float* abih    = (const float*)d_in[7];
  const float* abhh    = (const float*)d_in[8];
  const float* dec_wih = (const float*)d_in[9];
  const float* dec_whh = (const float*)d_in[10];
  const float* dbih    = (const float*)d_in[11];
  const float* dbhh    = (const float*)d_in[12];
  const float* proj_w  = (const float*)d_in[13];
  const float* proj_b  = (const float*)d_in[14];
  float* mel = (float*)d_out;

  char* p = (char*)d_ws;
  auto carve = [&](size_t bytes) -> char* {
    char* q = p; p += (bytes + 255) & ~(size_t)255; return q;
  };
  unsigned* sync  = (unsigned*)carve(256);
  float*    st    = (float*)carve((size_t)4 * BATCH * ATTD * sizeof(float));
  float *ah = st, *ac = st + BATCH * ATTD, *dh = st + 2 * BATCH * ATTD,
        *dc = st + 3 * BATCH * ATTD;
  __bf16* ahb = (__bf16*)carve((size_t)2 * BATCH * ATTD * 2);
  __bf16* dhb = (__bf16*)carve((size_t)2 * BATCH * ATTD * 2);
  __bf16* Wa  = (__bf16*)carve((size_t)4096 * KA * 2);
  __bf16* Wd  = (__bf16*)carve((size_t)4096 * KD * 2);
  __bf16* Wp  = (__bf16*)carve((size_t)160 * KP * 2);
  __bf16* pre = (__bf16*)carve((size_t)T_DEC * BATCH * PRED * 2);
  __bf16* ctx = (__bf16*)carve((size_t)T_DEC * BATCH * ED * 2);

  init_kernel<<<64, 256, 0, stream>>>(sync, st, ahb, dhb);
  pack2_bf16<<<2048, 256, 0, stream>>>(att_wih, 1024, att_whh, 1024, Wa, 4096);
  pack2_bf16<<<2048, 256, 0, stream>>>(dec_wih, 1792, dec_whh, 1024, Wd, 4096);
  pack2_bf16<<<256, 256, 0, stream>>>(proj_w, 1792, proj_w, 0, Wp, 160);
  ctx_kernel<<<2048, 256, 0, stream>>>(memory, ctx);
  prenet_kernel<<<T_DEC * BATCH, 256, 0, stream>>>(dec_in, w1, w2, pre);
  scan_kernel<<<NWG, TPB, 0, stream>>>(Wa, Wd, Wp, pre, ctx,
                                       abih, abhh, dbih, dbhh, proj_b,
                                       ah, ac, dh, dc, ahb, dhb, mel, sync);
}
